// TorchLMHeadGRPO_29429115912301
// MI455X (gfx1250) — matrix operand order repaired and checked by reading, not compiled, not run on hardware
//
#include <hip/hip_runtime.h>
#include <hip/hip_bf16.h>
#include <math.h>

// ---------------------------------------------------------------------------
// GRPO forward, fused LM-head. B=4, T=1024 (BT=4096), H=2048, V=32000.
// Compute-bound analysis: 2 GEMMs = 1.07 TFLOP >> 25us of HBM traffic, so
// minimize WMMA passes: single-pass FP16 WMMA (f32 accumulate), with a one-time
// fp32->f16 conversion pass so the GEMM hot loop does no conversion VALU.
// Pipeline (stream-ordered):
//   1. convert(x)->Xh, convert(W)->Wh          (fp32 -> _Float16)
//   2. gemm_stats_f16(Xh,Wh)  -> per (row, 256-col block) {max,argmax,sumexp,sumlogits}
//   3. row_reduce             -> per row {lse, chosen_lp=max-lse, chosen_idx, m2row}
//   4. convert(ref_x)->Xh, convert(ref_W)->Wh  (reuse buffers)
//   5. gemm_stats_f16 + row_reduce -> ref_lse
//   6. ref_chosen (fp32 dot at argmax) -> rowKL
//   7. final reduce -> d_out[4] = {loss, m1, m2, m3}
// Key identities: chosen logit == row max; coef_1 == 1 in fwd, so
// per_token_loss = -adv + BETA*kl;  m2 = mean_rows(sum_logits/V - lse).
// Workspace: Wh 131MB + Xh 16.8MB + partials 8.2MB + row arrays  (~156MB).
// ---------------------------------------------------------------------------

typedef __attribute__((ext_vector_type(16))) _Float16 v16h;
typedef __attribute__((ext_vector_type(8)))  float    v8f;

#define BT_   4096
#define HDIM  2048
#define VDIM  32000
#define NVB   125      // V / 256  (partial blocks per row)
#define XN    (4096 * 2048)
#define WN    (32000 * 2048)

union FragH { v16h v; uint4 q[2]; };   // 16 halves = 32B = 2 x b128 loads

// ---------------------------------------------------------------------------
// fp32 -> f16 conversion pass (vectorized, one-time cost per element)
// ---------------------------------------------------------------------------
__global__ __launch_bounds__(256) void convert_f16_kernel(
    const float* __restrict__ in, _Float16* __restrict__ out, int nvec)
{
  int i = blockIdx.x * 256 + threadIdx.x;
  if (i >= nvec) return;
  float4 v = ((const float4*)in)[i];
  union { _Float16 h[4]; uint2 u; } o;
  o.h[0] = (_Float16)v.x; o.h[1] = (_Float16)v.y;
  o.h[2] = (_Float16)v.z; o.h[3] = (_Float16)v.w;
  ((uint2*)out)[i] = o.u;
}

// ---------------------------------------------------------------------------
// GEMM + fused softmax partials. Wave tile: 32(M) x 64(N) = 8 WMMAs / k-step
// (independent accumulators -> no RAW chains). Workgroup (8 waves, 2x4):
// 64 rows x 256 cols. Fragments loaded as raw uint4 from the f16 arrays in the
// same contiguous k order (A 16x32 f16: lanes0-15 K0-15, lanes16-31
// K16-31; B 32x16: lanes0-15 K0-15, lanes16-31 K16-31; N=lane&15).
// ---------------------------------------------------------------------------
__global__ __launch_bounds__(256) void gemm_stats_f16(
    const _Float16* __restrict__ Xh, const _Float16* __restrict__ Wh,
    float* __restrict__ pmax, int* __restrict__ parg,
    float* __restrict__ psum, float* __restrict__ pslg)
{
  const int tid  = threadIdx.x;
  const int wave = tid >> 5, lane = tid & 31;
  const int hgrp = lane >> 4, n = lane & 15;
  const int wm   = wave >> 2, wn = wave & 3;

  const int row0 = blockIdx.x * 64 + wm * 32;
  const int col0 = blockIdx.y * 256 + wn * 64;

  const _Float16* ar0 = Xh + (size_t)(row0 +  0 + n) * HDIM;
  const _Float16* ar1 = Xh + (size_t)(row0 + 16 + n) * HDIM;
  const _Float16* br0 = Wh + (size_t)(col0 +  0 + n) * HDIM;
  const _Float16* br1 = Wh + (size_t)(col0 + 16 + n) * HDIM;
  const _Float16* br2 = Wh + (size_t)(col0 + 32 + n) * HDIM;
  const _Float16* br3 = Wh + (size_t)(col0 + 48 + n) * HDIM;

  const int kbase = hgrp * 16;   // A: lanes 16-31 hold K+16
  const int kb    = hgrp * 16;   // B: lanes 16-31 hold K+16

  v8f acc[2][4];
  #pragma unroll
  for (int mt = 0; mt < 2; ++mt)
    #pragma unroll
    for (int nt = 0; nt < 4; ++nt)
      acc[mt][nt] = (v8f){0.f,0.f,0.f,0.f,0.f,0.f,0.f,0.f};

  for (int kc = 0; kc < HDIM; kc += 32) {
    FragH a0, a1, b0, b1, b2, b3;
    a0.q[0] = *(const uint4*)(ar0 + kc + kbase);
    a0.q[1] = *(const uint4*)(ar0 + kc + kbase + 8);
    a1.q[0] = *(const uint4*)(ar1 + kc + kbase);
    a1.q[1] = *(const uint4*)(ar1 + kc + kbase + 8);
    b0.q[0] = *(const uint4*)(br0 + kc + kb);
    b0.q[1] = *(const uint4*)(br0 + kc + kb + 8);
    b1.q[0] = *(const uint4*)(br1 + kc + kb);
    b1.q[1] = *(const uint4*)(br1 + kc + kb + 8);
    b2.q[0] = *(const uint4*)(br2 + kc + kb);
    b2.q[1] = *(const uint4*)(br2 + kc + kb + 8);
    b3.q[0] = *(const uint4*)(br3 + kc + kb);
    b3.q[1] = *(const uint4*)(br3 + kc + kb + 8);

    acc[0][0] = __builtin_amdgcn_wmma_f32_16x16x32_f16(false, a0.v, false, b0.v, (short)0, acc[0][0], false, false);
    acc[0][1] = __builtin_amdgcn_wmma_f32_16x16x32_f16(false, a0.v, false, b1.v, (short)0, acc[0][1], false, false);
    acc[0][2] = __builtin_amdgcn_wmma_f32_16x16x32_f16(false, a0.v, false, b2.v, (short)0, acc[0][2], false, false);
    acc[0][3] = __builtin_amdgcn_wmma_f32_16x16x32_f16(false, a0.v, false, b3.v, (short)0, acc[0][3], false, false);
    acc[1][0] = __builtin_amdgcn_wmma_f32_16x16x32_f16(false, a1.v, false, b0.v, (short)0, acc[1][0], false, false);
    acc[1][1] = __builtin_amdgcn_wmma_f32_16x16x32_f16(false, a1.v, false, b1.v, (short)0, acc[1][1], false, false);
    acc[1][2] = __builtin_amdgcn_wmma_f32_16x16x32_f16(false, a1.v, false, b2.v, (short)0, acc[1][2], false, false);
    acc[1][3] = __builtin_amdgcn_wmma_f32_16x16x32_f16(false, a1.v, false, b3.v, (short)0, acc[1][3], false, false);
  }

  // --- fused per-row softmax partials over this workgroup's 256 columns ---
  __shared__ float smax[64][4];
  __shared__ int   sarg[64][4];
  __shared__ float ssum[64][4];
  __shared__ float sslg[64][4];

  #pragma unroll
  for (int mt = 0; mt < 2; ++mt) {
    #pragma unroll
    for (int r = 0; r < 8; ++r) {
      float v0 = acc[mt][0][r], v1 = acc[mt][1][r];
      float v2 = acc[mt][2][r], v3 = acc[mt][3][r];
      // merge 4 N-subtiles (ascending col index; strict > keeps first max)
      float mx = v0; int ag = col0 + n;
      if (v1 > mx) { mx = v1; ag = col0 + 16 + n; }
      if (v2 > mx) { mx = v2; ag = col0 + 32 + n; }
      if (v3 > mx) { mx = v3; ag = col0 + 48 + n; }
      #pragma unroll
      for (int msk = 1; msk <= 8; msk <<= 1) {   // within 16-lane half-wave
        float om = __shfl_xor(mx, msk, 32);
        int   oa = __shfl_xor(ag, msk, 32);
        if (om > mx || (om == mx && oa < ag)) { mx = om; ag = oa; }
      }
      float sl = v0 + v1 + v2 + v3;
      float e  = expf(v0 - mx) + expf(v1 - mx) + expf(v2 - mx) + expf(v3 - mx);
      #pragma unroll
      for (int msk = 1; msk <= 8; msk <<= 1) {
        sl += __shfl_xor(sl, msk, 32);
        e  += __shfl_xor(e,  msk, 32);
      }
      if (n == 0) {
        int rr = wm * 32 + mt * 16 + hgrp * 8 + r;    // row within workgroup
        smax[rr][wn] = mx; sarg[rr][wn] = ag;
        ssum[rr][wn] = e;  sslg[rr][wn] = sl;
      }
    }
  }
  __syncthreads();

  if (tid < 64) {
    float bm = smax[tid][0]; int ba = sarg[tid][0];
    #pragma unroll
    for (int w = 1; w < 4; ++w) {
      float m = smax[tid][w]; int a = sarg[tid][w];
      if (m > bm || (m == bm && a < ba)) { bm = m; ba = a; }
    }
    float bs = 0.f, bl = 0.f;
    #pragma unroll
    for (int w = 0; w < 4; ++w) {
      bs += ssum[tid][w] * expf(smax[tid][w] - bm);
      bl += sslg[tid][w];
    }
    size_t idx = (size_t)(blockIdx.x * 64 + tid) * NVB + blockIdx.y;
    pmax[idx] = bm; parg[idx] = ba; psum[idx] = bs; pslg[idx] = bl;
  }
}

// ---------------------------------------------------------------------------
// One wave per row: merge 125 partials (online logsumexp merge).
// ---------------------------------------------------------------------------
__global__ __launch_bounds__(256) void row_reduce_kernel(
    const float* __restrict__ pmax, const int* __restrict__ parg,
    const float* __restrict__ psum, const float* __restrict__ pslg,
    float* __restrict__ lse_out, float* __restrict__ clp_out,
    int* __restrict__ idx_out, float* __restrict__ m2_out)
{
  const int wave = threadIdx.x >> 5, lane = threadIdx.x & 31;
  const int row  = blockIdx.x * 8 + wave;
  const float* pm = pmax + (size_t)row * NVB;
  const int*   pa = parg + (size_t)row * NVB;
  const float* ps = psum + (size_t)row * NVB;
  const float* pl = pslg + (size_t)row * NVB;

  float m = -INFINITY, s = 0.f, sl = 0.f; int a = 0x7fffffff;
  for (int i = lane; i < NVB; i += 32) {
    float mi = pm[i], si = ps[i]; int ai = pa[i];
    sl += pl[i];
    if (mi > m || (mi == m && ai < a)) { s = s * expf(m - mi) + si; m = mi; a = ai; }
    else                               { s += si * expf(mi - m); }
  }
  #pragma unroll
  for (int msk = 1; msk <= 16; msk <<= 1) {
    float om = __shfl_xor(m, msk, 32);
    int   oa = __shfl_xor(a, msk, 32);
    float os = __shfl_xor(s, msk, 32);
    sl += __shfl_xor(sl, msk, 32);
    if (om > m || (om == m && oa < a)) { s = s * expf(m - om) + os; m = om; a = oa; }
    else                               { s += os * expf(om - m); }
  }
  if (lane == 0) {
    float lse = m + logf(s);
    lse_out[row] = lse;
    clp_out[row] = m - lse;        // chosen logit == row max
    idx_out[row] = a;
    m2_out[row]  = sl / (float)VDIM - lse;
  }
}

// ---------------------------------------------------------------------------
// ref_logit at chosen index (fp32 2048-dot per row) + KL per row.
// ---------------------------------------------------------------------------
__global__ __launch_bounds__(256) void ref_chosen_kernel(
    const float* __restrict__ refX, const float* __restrict__ refW,
    const int* __restrict__ chosen, const float* __restrict__ clp,
    const float* __restrict__ ref_lse, float* __restrict__ rowKL)
{
  const int wave = threadIdx.x >> 5, lane = threadIdx.x & 31;
  const int row  = blockIdx.x * 8 + wave;
  const float* xr = refX + (size_t)row * HDIM;
  const float* wr = refW + (size_t)chosen[row] * HDIM;
  float acc = 0.f;
  for (int k = lane * 4; k < HDIM; k += 128) {
    float4 av = *(const float4*)(xr + k);
    float4 bv = *(const float4*)(wr + k);
    acc += av.x * bv.x + av.y * bv.y + av.z * bv.z + av.w * bv.w;
  }
  #pragma unroll
  for (int msk = 1; msk <= 16; msk <<= 1) acc += __shfl_xor(acc, msk, 32);
  if (lane == 0) {
    float ref_lp = acc - ref_lse[row];
    float delta  = ref_lp - clp[row];
    rowKL[row]   = expf(delta) - delta - 1.f;
  }
}

// ---------------------------------------------------------------------------
// Final reduction over 4096 rows -> {loss, m1, m2, m3}.
// ---------------------------------------------------------------------------
__device__ __forceinline__ float block_reduce_1024(float v, float* sbuf) {
  int t = threadIdx.x;
  sbuf[t] = v; __syncthreads();
  for (int s = 512; s > 0; s >>= 1) {
    if (t < s) sbuf[t] += sbuf[t + s];
    __syncthreads();
  }
  float r = sbuf[0]; __syncthreads();
  return r;
}

__global__ __launch_bounds__(1024) void final_kernel(
    const float* __restrict__ attn, const float* __restrict__ adv,
    const float* __restrict__ clp, const float* __restrict__ m2row,
    const float* __restrict__ rowKL, float* __restrict__ out)
{
  __shared__ float sbuf[1024];
  const int t = threadIdx.x;
  float lossnum = 0.f, masksum = 0.f, m1 = 0.f, m2 = 0.f;
  float klm[4], mb[4];
  #pragma unroll
  for (int b = 0; b < 4; ++b) {              // T == 1024 == blockDim
    int row    = b * 1024 + t;
    float mask = attn[row];
    float kl   = rowKL[row];
    lossnum   += (-adv[b] + 0.1f * kl) * mask;   // BETA = 0.1, coef_1 == 1
    masksum   += mask;
    m1        += clp[row];
    m2        += m2row[row];
    klm[b] = kl * mask;
    mb[b]  = mask;
  }
  lossnum = block_reduce_1024(lossnum, sbuf);
  masksum = block_reduce_1024(masksum, sbuf);
  m1      = block_reduce_1024(m1, sbuf);
  m2      = block_reduce_1024(m2, sbuf);
  float m3 = 0.f;
  #pragma unroll
  for (int b = 0; b < 4; ++b) {
    float k  = block_reduce_1024(klm[b], sbuf);
    float mm = block_reduce_1024(mb[b], sbuf);
    if (t == 0) m3 += k / fmaxf(mm, 1.f);
  }
  if (t == 0) {
    out[0] = lossnum / fmaxf(masksum, 1.f);
    out[1] = m1 / (float)BT_;
    out[2] = m2 / (float)BT_;
    out[3] = m3 * 0.25f;
  }
}

// ---------------------------------------------------------------------------
extern "C" void kernel_launch(void* const* d_in, const int* in_sizes, int n_in,
                              void* d_out, int out_size, void* d_ws, size_t ws_size,
                              hipStream_t stream)
{
  (void)in_sizes; (void)n_in; (void)out_size; (void)ws_size;
  const float* X    = (const float*)d_in[0];   // (4,1024,2048)
  const float* attn = (const float*)d_in[1];   // (4,1024)
  const float* adv  = (const float*)d_in[2];   // (4,)
  const float* refX = (const float*)d_in[3];   // (4,1024,2048)
  const float* W    = (const float*)d_in[4];   // (32000,2048)
  const float* refW = (const float*)d_in[5];   // (32000,2048)

  // workspace carve-out (~156 MB); f16 + partial buffers reused for main & ref
  char* ws = (char*)d_ws;
  _Float16* Wh = (_Float16*)ws; ws += (size_t)WN * 2;      // 131 MB
  _Float16* Xh = (_Float16*)ws; ws += (size_t)XN * 2;      // 16.8 MB
  const size_t P = (size_t)BT_ * NVB;
  float* pmax = (float*)ws; ws += P * 4;
  int*   parg = (int*)ws;   ws += P * 4;
  float* psum = (float*)ws; ws += P * 4;
  float* pslg = (float*)ws; ws += P * 4;
  float* clp     = (float*)ws; ws += BT_ * 4;
  int*   cidx    = (int*)ws;   ws += BT_ * 4;
  float* m2row   = (float*)ws; ws += BT_ * 4;
  float* lse_m   = (float*)ws; ws += BT_ * 4;
  float* ref_lse = (float*)ws; ws += BT_ * 4;
  float* scr_clp = (float*)ws; ws += BT_ * 4;
  int*   scr_idx = (int*)ws;   ws += BT_ * 4;
  float* scr_m2  = (float*)ws; ws += BT_ * 4;
  float* rowKL   = (float*)ws; ws += BT_ * 4;

  const int xvec = XN / 4, wvec = WN / 4;
  dim3 blk(256);
  dim3 gGemm(BT_ / 64, NVB);     // row blocks fast => W stripe reuse in L2

  // main policy GEMM
  convert_f16_kernel<<<dim3((xvec + 255) / 256), blk, 0, stream>>>(X, Xh, xvec);
  convert_f16_kernel<<<dim3((wvec + 255) / 256), blk, 0, stream>>>(W, Wh, wvec);
  gemm_stats_f16<<<gGemm, blk, 0, stream>>>(Xh, Wh, pmax, parg, psum, pslg);
  row_reduce_kernel<<<dim3(BT_ / 8), blk, 0, stream>>>(
      pmax, parg, psum, pslg, lse_m, clp, cidx, m2row);

  // reference GEMM (reuse f16 + partial buffers)
  convert_f16_kernel<<<dim3((xvec + 255) / 256), blk, 0, stream>>>(refX, Xh, xvec);
  convert_f16_kernel<<<dim3((wvec + 255) / 256), blk, 0, stream>>>(refW, Wh, wvec);
  gemm_stats_f16<<<gGemm, blk, 0, stream>>>(Xh, Wh, pmax, parg, psum, pslg);
  row_reduce_kernel<<<dim3(BT_ / 8), blk, 0, stream>>>(
      pmax, parg, psum, pslg, ref_lse, scr_clp, scr_idx, scr_m2);

  ref_chosen_kernel<<<dim3(BT_ / 8), blk, 0, stream>>>(
      refX, refW, cidx, clp, ref_lse, rowKL);
  final_kernel<<<1, 1024, 0, stream>>>(attn, adv, clp, m2row, rowKL, (float*)d_out);
}
